// VanillaAttention_34737695490632
// MI455X (gfx1250) — matrix operand order repaired and checked by reading, not compiled, not run on hardware
//
#include <hip/hip_runtime.h>
#include <hip/hip_bf16.h>

// ---------------------------------------------------------------------------
// Bahdanau additive attention for MI455X (gfx1250, wave32).
//   qp = q@Wq+bq ; kp = k@Wk+bk                      (f32 WMMA 16x16x4)
//   score[n,m] = b + sum_d w_d * sigmoid(qp+kp)      (v_tanh_f32 bound)
//   out = softmax(score) @ v                         (bf16 WMMA 16x16x32)
// d_out = [ output (2048*64 f32) | att_score (2048*2048 f32) ]
// ---------------------------------------------------------------------------

#define N1 2048
#define N2 2048
#define DD 64
#define L2E 1.4426950408889634f

typedef __attribute__((ext_vector_type(2)))  float  v2f;
typedef __attribute__((ext_vector_type(8)))  float  v8f;
typedef __attribute__((ext_vector_type(16))) __bf16 v16bf;

#if defined(__has_builtin)
#if __has_builtin(__builtin_amdgcn_global_load_async_to_lds_b32)
#define HAVE_ASYNC_LDS 1
#endif
#endif

#if defined(HAVE_ASYNC_LDS)
typedef __attribute__((address_space(1))) int gas_int;  // global (prints as __device__ int*)
typedef __attribute__((address_space(3))) int las_int;  // LDS
#endif

// ---------------------------------------------------------------------------
// Kernel 1: Y_half = 0.5*(X @ W + bias), one 16x16 tile per wave via
// V_WMMA_F32_16X16X4_F32.  X:[rows,64], W:[64,64] row-major.
// A 16x4 f32 layout: lanes 0-15 row M=lane, K={k,k+1}; lanes 16-31 K={k+2,k+3}.
// B 4x16 f32 layout: lanes hold column N=lane%16, K pair analogous.
// C/D: VGPR r -> M = r + (lane<16 ? 0 : 8), N = lane%16.
// ---------------------------------------------------------------------------
__global__ void proj_half_wmma(const float* __restrict__ X,
                               const float* __restrict__ W,
                               const float* __restrict__ bias,
                               float* __restrict__ Yh) {
  const int wave = blockIdx.x * (blockDim.x >> 5) + (threadIdx.x >> 5);
  const int lane = threadIdx.x & 31;
  const int mTile = wave >> 2;        // 128 row tiles
  const int nTile = wave & 3;         // 4 col tiles
  const bool laneLo = (lane < 16);
  const int r = lane & 15;
  const int kOff = laneLo ? 0 : 2;

  const float* __restrict__ Arow = X + (mTile * 16 + r) * DD;
  const int col = nTile * 16 + r;

  v8f c = {};
#pragma unroll
  for (int k = 0; k < DD; k += 4) {
    v2f a, bf;
    a.x  = Arow[k + kOff];
    a.y  = Arow[k + kOff + 1];
    bf.x = W[(k + kOff) * DD + col];
    bf.y = W[(k + kOff + 1) * DD + col];
    c = __builtin_amdgcn_wmma_f32_16x16x4_f32(false, a, false, bf,
                                              (short)0, c, false, false);
  }
  const float bc = bias[col];
#pragma unroll
  for (int i = 0; i < 8; ++i) {
    const int row = mTile * 16 + i + (laneLo ? 0 : 8);
    Yh[row * DD + col] = 0.5f * (c[i] + bc);
  }
}

// ---------------------------------------------------------------------------
// Kernel 2: score[n,m] = b + sum_d w_d*sigmoid(qp+kp)
//         = (b + sum_d w_d/2) + sum_d (w_d/2)*tanh(qh+kh),  qh=qp/2, kh=kp/2.
// 64x64 tile per 256-thread block; 4x4 register tile per thread.
// Tile fill uses CDNA5 async global->LDS copies (ASYNCcnt) when available.
// Dominant cost: 64 v_tanh_f32 per output element.
// ---------------------------------------------------------------------------
__global__ void score_kernel(const float* __restrict__ qh,
                             const float* __restrict__ kh,
                             const float* __restrict__ w,
                             const float* __restrict__ bptr,
                             float* __restrict__ score) {
  __shared__ float qs[64][65];   // pad stride 65 -> conflict-free d-column reads
  __shared__ float ks[64][65];
  __shared__ float wh[64];

  const int tid = threadIdx.x;
  const int nT = blockIdx.y * 64;
  const int mT = blockIdx.x * 64;

#if defined(HAVE_ASYNC_LDS)
  // Cache->LDS DMA, 4B granularity (padded rows are not 16B aligned).
  for (int idx = tid; idx < 64 * DD; idx += 256) {
    const int row = idx >> 6, colD = idx & 63;
    __builtin_amdgcn_global_load_async_to_lds_b32(
        (gas_int*)(qh + (size_t)(nT + row) * DD + colD),
        (las_int*)&qs[row][colD], 0, 0);
    __builtin_amdgcn_global_load_async_to_lds_b32(
        (gas_int*)(kh + (size_t)(mT + row) * DD + colD),
        (las_int*)&ks[row][colD], 0, 0);
  }
  if (tid < 64) wh[tid] = 0.5f * w[tid];
#if __has_builtin(__builtin_amdgcn_s_wait_asynccnt)
  __builtin_amdgcn_s_wait_asynccnt(0);
#else
  asm volatile("s_wait_asynccnt 0x0" ::: "memory");
#endif
#else
  for (int idx = tid; idx < 64 * DD; idx += 256) {
    const int row = idx >> 6, colD = idx & 63;
    qs[row][colD] = qh[(nT + row) * DD + colD];
    ks[row][colD] = kh[(mT + row) * DD + colD];
  }
  if (tid < 64) wh[tid] = 0.5f * w[tid];
#endif
  __syncthreads();

  float cbase = bptr[0];
#if __has_builtin(__builtin_amdgcn_tanhf)
  for (int d = 0; d < DD; ++d) cbase += wh[d];   // b + sum w/2 (LDS broadcast)
#endif

  const int tx = tid & 15;        // m sub-tile
  const int ty = tid >> 4;        // n sub-tile
  float acc[4][4];
#pragma unroll
  for (int i = 0; i < 4; ++i)
#pragma unroll
    for (int j = 0; j < 4; ++j) acc[i][j] = cbase;

  for (int d = 0; d < DD; ++d) {
    const float wd = wh[d];
    float qv[4], kv[4];
#pragma unroll
    for (int i = 0; i < 4; ++i) qv[i] = qs[ty * 4 + i][d];
#pragma unroll
    for (int j = 0; j < 4; ++j) kv[j] = ks[tx * 4 + j][d];
#pragma unroll
    for (int i = 0; i < 4; ++i)
#pragma unroll
      for (int j = 0; j < 4; ++j) {
        const float z = qv[i] + kv[j];
#if __has_builtin(__builtin_amdgcn_tanhf)
        acc[i][j] += wd * __builtin_amdgcn_tanhf(z);          // 1 trans/term
#else
        // sigma(2z) = 1/(1+exp2(-2z*log2e)); w = 2*wh
        acc[i][j] += (2.0f * wd) *
            __builtin_amdgcn_rcpf(1.0f +
                __builtin_amdgcn_exp2f(z * (-2.0f * L2E)));   // 2 trans/term
#endif
      }
  }

#pragma unroll
  for (int i = 0; i < 4; ++i) {
    float4 o = make_float4(acc[i][0], acc[i][1], acc[i][2], acc[i][3]);
    *(float4*)&score[(nT + ty * 4 + i) * N2 + mT + tx * 4] = o;
  }
}

// ---------------------------------------------------------------------------
// Kernel 3: per-row softmax statistics: rmax[n], rinv[n] = 1/sum exp(s-max).
// One 256-thread block per row.
// ---------------------------------------------------------------------------
__global__ void rowstat_kernel(const float* __restrict__ score,
                               float* __restrict__ rmax,
                               float* __restrict__ rinv) {
  __shared__ float red[256];
  const int row = blockIdx.x;
  const int tid = threadIdx.x;
  const float* __restrict__ s = score + row * (size_t)N2;

  float m = -3.4e38f;
  for (int i = tid; i < N2; i += 256) m = fmaxf(m, s[i]);
  red[tid] = m;
  __syncthreads();
  for (int off = 128; off > 0; off >>= 1) {
    if (tid < off) red[tid] = fmaxf(red[tid], red[tid + off]);
    __syncthreads();
  }
  m = red[0];
  __syncthreads();

  float sum = 0.0f;
  for (int i = tid; i < N2; i += 256)
    sum += __builtin_amdgcn_exp2f((s[i] - m) * L2E);
  red[tid] = sum;
  __syncthreads();
  for (int off = 128; off > 0; off >>= 1) {
    if (tid < off) red[tid] += red[tid + off];
    __syncthreads();
  }
  if (tid == 0) {
    rmax[row] = m;
    rinv[row] = 1.0f / red[0];
  }
}

// ---------------------------------------------------------------------------
// Kernel 4: convert v to bf16 once (reused by 128 row-tiles of the P@V GEMM).
// ---------------------------------------------------------------------------
__global__ void cvt_bf16_kernel(const float* __restrict__ v,
                                __bf16* __restrict__ vb) {
  const int i = blockIdx.x * blockDim.x + threadIdx.x;
  vb[i] = (__bf16)v[i];
}

// ---------------------------------------------------------------------------
// Kernel 5: out = softmax(score) @ v via V_WMMA_F32_16X16X32_BF16.
// Probabilities built on the fly: p = exp2((s-rmax)*log2e)*rinv, cast bf16.
// A 16x32 bf16 layout (per lane row M=lane%16):
//   VGPR v<4 -> K=2v{+1}, v>=4 -> K=16+2(v-4){+1}; +8 for lanes 16-31.
// B 32x16 bf16 layout (per lane col N=lane%16):
//   same K mapping as A (v<4 -> K=2v{+1}, v>=4 -> K=16+2(v-4){+1}; +8 for lanes 16-31).
// Next k-step's operands are prefetched (global_prefetch_b8) while the
// current step's exp2 + WMMA execute.
// ---------------------------------------------------------------------------
__global__ void attn_out_wmma(const float* __restrict__ score,
                              const float* __restrict__ rmax,
                              const float* __restrict__ rinv,
                              const __bf16* __restrict__ vb,
                              float* __restrict__ out) {
  const int wave = blockIdx.x * (blockDim.x >> 5) + (threadIdx.x >> 5);
  const int lane = threadIdx.x & 31;
  const int mTile = wave >> 2;        // 128 n tiles
  const int dTile = wave & 3;         // 4 d tiles
  const bool laneLo = (lane < 16);
  const int r = lane & 15;

  const int row = mTile * 16 + r;                       // fixed score row/lane
  const float* __restrict__ srow = score + row * (size_t)N2;
  const float mx = rmax[row];
  const float inv = rinv[row];
  const int aBase = laneLo ? 0 : 8;
  const int bBase = laneLo ? 0 : 16;
  const int dcol = dTile * 16 + r;

  v8f c = {};
  for (int k = 0; k < N2; k += 32) {
    if (k + 32 < N2) {
      __builtin_prefetch(&srow[k + 32 + aBase], 0, 1);       // next A chunk
      __builtin_prefetch(&vb[(k + 32 + bBase) * DD + dcol], 0, 1); // next B
    }
    v16bf a, b;
#pragma unroll
    for (int e = 0; e < 16; ++e) {
      const int vr = e >> 1, half = e & 1;
      const int Ka = (vr < 4 ? 2 * vr : 16 + 2 * (vr - 4)) + aBase + half;
      const float p = __builtin_amdgcn_exp2f((srow[k + Ka] - mx) * L2E) * inv;
      a[e] = (__bf16)p;
      const int Kb = (vr < 4 ? 2 * vr : 16 + 2 * (vr - 4)) + aBase + half;
      b[e] = vb[(k + Kb) * DD + dcol];
    }
    c = __builtin_amdgcn_wmma_f32_16x16x32_bf16(false, a, false, b,
                                                (short)0, c, false, false);
  }

#pragma unroll
  for (int i = 0; i < 8; ++i) {
    const int orow = mTile * 16 + i + (laneLo ? 0 : 8);
    out[orow * DD + dcol] = c[i];
  }
}

// ---------------------------------------------------------------------------
extern "C" void kernel_launch(void* const* d_in, const int* in_sizes, int n_in,
                              void* d_out, int out_size, void* d_ws, size_t ws_size,
                              hipStream_t stream) {
  const float* q  = (const float*)d_in[0];
  const float* k  = (const float*)d_in[1];
  const float* v  = (const float*)d_in[2];
  const float* Wq = (const float*)d_in[3];
  const float* bq = (const float*)d_in[4];
  const float* Wk = (const float*)d_in[5];
  const float* bk = (const float*)d_in[6];
  const float* w  = (const float*)d_in[7];
  const float* b  = (const float*)d_in[8];

  float* out = (float*)d_out;               // [N1, D]
  float* att = out + (size_t)N1 * DD;       // [N1, N2] (second tuple output)

  // workspace layout
  float*  qh   = (float*)d_ws;              // N1*D halved q-projection
  float*  kh   = qh + (size_t)N1 * DD;      // N2*D halved k-projection
  float*  rmx  = kh + (size_t)N2 * DD;      // N1 row maxima
  float*  rinv = rmx + N1;                  // N1 reciprocal row sums
  __bf16* vb   = (__bf16*)(rinv + N1);      // N2*D bf16 copy of v

  // 1) projections: 512 wave-tiles each, 8 waves/block -> 64 blocks
  proj_half_wmma<<<64, 256, 0, stream>>>(q, Wq, bq, qh);
  proj_half_wmma<<<64, 256, 0, stream>>>(k, Wk, bk, kh);

  // 2) v -> bf16 (N2*D = 131072 elements)
  cvt_bf16_kernel<<<(N2 * DD) / 256, 256, 0, stream>>>(v, vb);

  // 3) scores: 32x32 tiles of 64x64
  dim3 sg(N2 / 64, N1 / 64);
  score_kernel<<<sg, 256, 0, stream>>>(qh, kh, w, b, att);

  // 4) softmax row statistics
  rowstat_kernel<<<N1, 256, 0, stream>>>(att, rmx, rinv);

  // 5) P @ V: 512 wave-tiles, 8 waves/block -> 64 blocks
  attn_out_wmma<<<64, 256, 0, stream>>>(att, rmx, rinv, vb, out);
}
